// GroupedQueryAttention_31035433681366
// MI455X (gfx1250) — matrix operand order repaired and checked by reading, not compiled, not run on hardware
//
#include <hip/hip_runtime.h>
#include <hip/hip_bf16.h>

// ---------------------------------------------------------------------------
// CDNA5 (gfx1250) GQA block: bf16 WMMA GEMMs (TDM double-buffered W tiles) +
// flash attention, f32 accumulation throughout.
// ---------------------------------------------------------------------------

typedef __attribute__((ext_vector_type(16))) __bf16 v16bf;
typedef __attribute__((ext_vector_type(8)))  float  v8f;

union B32x8 { uint4 u4[2]; v16bf v; };

__device__ __forceinline__ v8f wmma_bf16(v16bf a, v16bf b, v8f c) {
  // D = A(16x32 bf16) x B(32x16 bf16) + C(16x16 f32)
  return __builtin_amdgcn_wmma_f32_16x16x32_bf16(
      /*neg_a=*/false, a, /*neg_b=*/false, b,
      /*c_mod=*/(short)0, c, /*reuse_a=*/false, /*reuse_b=*/false);
}

// A fragment (16x32, 16-bit): lane hf=lane/16, idx=lane%16 holds row M=idx,
// K slots {hf*8..hf*8+7} and {16+hf*8..16+hf*8+7}  (ISA 7.12.2)
__device__ __forceinline__ v16bf load_a_frag(const __bf16* p, int ld) {
  int lane = threadIdx.x & 31;
  int hf = lane >> 4, idx = lane & 15;
  const __bf16* row = p + (size_t)idx * ld + hf * 8;
  B32x8 t;
  t.u4[0] = *reinterpret_cast<const uint4*>(row);       // K = hf*8 + 0..7
  t.u4[1] = *reinterpret_cast<const uint4*>(row + 16);  // K = 16+hf*8 + 0..7
  return t.v;
}

// B fragment (32x16, 16-bit), source stored as B^T rows (N x K row-major):
// lane holds column N=idx, K slots {hf*8..hf*8+7} and {16+hf*8..16+hf*8+7}
__device__ __forceinline__ v16bf load_b_frag(const __bf16* p, int ld) {
  int lane = threadIdx.x & 31;
  int hf = lane >> 4, idx = lane & 15;
  const __bf16* row = p + (size_t)idx * ld + hf * 8;
  B32x8 t;
  t.u4[0] = *reinterpret_cast<const uint4*>(row);
  t.u4[1] = *reinterpret_cast<const uint4*>(row + 16);
  return t.v;
}

// ---------------------------------------------------------------------------
// Tensor Data Mover: 2-D tile load (rows x cols bf16) global -> LDS.
#if __has_builtin(__builtin_amdgcn_tensor_load_to_lds)
#define HAVE_TDM 1
typedef unsigned int u32x4 __attribute__((ext_vector_type(4)));
typedef int i32x4 __attribute__((ext_vector_type(4)));
typedef int i32x8 __attribute__((ext_vector_type(8)));

__device__ __forceinline__ void tdm_load_2d(const __bf16* gsrc,
                                            unsigned lds_off, int rows,
                                            int cols, int ld_elems) {
  unsigned long long ga = (unsigned long long)(const void*)gsrc;
  u32x4 g0;
  g0[0] = 1u;                                   // count=1 (valid descriptor)
  g0[1] = lds_off;                              // LDS byte address
  g0[2] = (unsigned)(ga & 0xffffffffu);         // global_addr[31:0] @bit64
  g0[3] = (unsigned)((ga >> 32) & 0x01ffffffu)  // global_addr[56:32]
          | (2u << 30);                         // type = 2 ("image")
  unsigned w[8] = {};
  w[0] = 1u << 16;                                     // data_size = 2 bytes
  w[1] = ((unsigned)cols & 0xffffu) << 16;             // tensor_dim0 @bit48
  w[2] = (((unsigned)cols >> 16) & 0xffffu)            // tensor_dim0 hi
         | (((unsigned)rows & 0xffffu) << 16);         // tensor_dim1 @bit80
  w[3] = (((unsigned)rows >> 16) & 0xffffu)            // tensor_dim1 hi
         | (((unsigned)cols & 0xffffu) << 16);         // tile_dim0 @bit112
  w[4] = (unsigned)rows & 0xffffu;                     // tile_dim1 @bit128
  w[5] = (unsigned)ld_elems;                           // tensor_dim0_stride lo
  i32x8 g1;
#pragma unroll
  for (int i = 0; i < 8; ++i) g1[i] = (int)w[i];
  i32x4 gz = {0, 0, 0, 0};
#if defined(__has_include)
#if __has_include(<hip/amd_detail/amd_gfx1250_TDM.h>)
  i32x8 gz8 = {};
  __builtin_amdgcn_tensor_load_to_lds(g0, g1, gz, gz, gz8, 0);  // clang-23 form
#else
  __builtin_amdgcn_tensor_load_to_lds(g0, g1, gz, gz, 0);       // ROCm 7.2 form
#endif
#else
  __builtin_amdgcn_tensor_load_to_lds(g0, g1, gz, gz, 0);
#endif
}

__device__ __forceinline__ void tdm_wait0() {
#if __has_builtin(__builtin_amdgcn_s_wait_tensorcnt)
  __builtin_amdgcn_s_wait_tensorcnt(0);
#else
  asm volatile("s_wait_tensorcnt 0x0" ::: "memory");
#endif
}
__device__ __forceinline__ void tdm_wait1() {
#if __has_builtin(__builtin_amdgcn_s_wait_tensorcnt)
  __builtin_amdgcn_s_wait_tensorcnt(1);
#else
  asm volatile("s_wait_tensorcnt 0x1" ::: "memory");
#endif
}
#else
#define HAVE_TDM 0
#endif

// ---------------------------------------------------------------------------
__global__ void cast_f32_bf16(const float* __restrict__ in,
                              __bf16* __restrict__ out, int n) {
  int i = blockIdx.x * blockDim.x + threadIdx.x;
  if (i < n) out[i] = (__bf16)in[i];
}

// V (B*T, 512) -> Vt (B*4, 128, T): makes P@V B-fragments contiguous.
__global__ void transpose_v(const __bf16* __restrict__ V,
                            __bf16* __restrict__ Vt, int T, int total) {
  int i = blockIdx.x * blockDim.x + threadIdx.x;
  if (i >= total) return;
  int t = i % T;
  int d = (i / T) & 127;
  int khb = i / (T * 128);  // b*4 + kh
  int b = khb >> 2, kh = khb & 3;
  Vt[i] = V[((size_t)b * T + t) * 512 + kh * 128 + d];
}

// ---------------------------------------------------------------------------
// Y[M,N] = X[M,K] @ W[N,K]^T ; bf16 inputs, f32 accumulate.
// Block: 256 threads (8 waves). Block tile 128(M) x 128(N); wave -> 32x64.
// W tile (128x32 bf16 = 8KB) staged in LDS: TDM double-buffered when
// available, cooperative b128 loads otherwise.
template <bool OUT_F32>
__global__ __launch_bounds__(256) void gemm_xwT(const __bf16* __restrict__ X,
                                                const __bf16* __restrict__ W,
                                                void* __restrict__ Y,
                                                int M, int N, int K) {
  __shared__ __bf16 lB[2][128 * 32];
  int wave = threadIdx.x >> 5;
  int lane = threadIdx.x & 31;
  int hf = lane >> 4, idx = lane & 15;
  int wm = wave & 3;   // 4 M-strips of 32
  int wn = wave >> 2;  // 2 N-strips of 64
  int row0 = blockIdx.x * 128 + wm * 32;
  int colB = blockIdx.y * 128;

  v8f acc[2][4] = {};
  int nsteps = K / 32;

#if HAVE_TDM
  if (threadIdx.x < 32)
    tdm_load_2d(&W[(size_t)colB * K], (unsigned)(size_t)&lB[0][0], 128, 32, K);
#endif

  for (int s = 0; s < nsteps; ++s) {
    int k0 = s * 32;
#if HAVE_TDM
    if (threadIdx.x < 32) {
      if (s + 1 < nsteps) {
        tdm_load_2d(&W[(size_t)colB * K + (k0 + 32)],
                    (unsigned)(size_t)&lB[(s + 1) & 1][0], 128, 32, K);
        tdm_wait1();  // tile s done; tile s+1 in flight
      } else {
        tdm_wait0();
      }
    }
    __syncthreads();
    const __bf16* Bt = &lB[s & 1][0];
#else
    {
      int t = threadIdx.x;
      int r = t >> 1;            // 0..127
      int c = (t & 1) * 16;      // 0 or 16
      const uint4* src =
          reinterpret_cast<const uint4*>(&W[(size_t)(colB + r) * K + k0 + c]);
      uint4* dst = reinterpret_cast<uint4*>(&lB[0][r * 32 + c]);
      dst[0] = src[0];
      dst[1] = src[1];
    }
    __syncthreads();
    const __bf16* Bt = &lB[0][0];
#endif
    v16bf a0 = load_a_frag(X + (size_t)row0 * K + k0, K);
    v16bf a1 = load_a_frag(X + (size_t)(row0 + 16) * K + k0, K);
#pragma unroll
    for (int f = 0; f < 4; ++f) {
      v16bf b = load_b_frag(Bt + (size_t)(wn * 64 + f * 16) * 32, 32);
      acc[0][f] = wmma_bf16(a0, b, acc[0][f]);
      acc[1][f] = wmma_bf16(a1, b, acc[1][f]);
    }
    __syncthreads();
  }

#pragma unroll
  for (int mi = 0; mi < 2; ++mi)
#pragma unroll
    for (int f = 0; f < 4; ++f)
#pragma unroll
      for (int i = 0; i < 8; ++i) {
        size_t r = (size_t)(row0 + mi * 16 + i + hf * 8);
        size_t c = (size_t)(colB + wn * 64 + f * 16 + idx);
        if (OUT_F32)
          reinterpret_cast<float*>(Y)[r * N + c] = acc[mi][f][i];
        else
          reinterpret_cast<__bf16*>(Y)[r * N + c] = (__bf16)acc[mi][f][i];
      }
}

// ---------------------------------------------------------------------------
// In-place RoPE on bf16 Q or K, layout (B*T, heads*128).
// freqs: 32 values 10000^(-j/31), then 32 zeros (cos=1,sin=0).
__global__ void rope_inplace(__bf16* __restrict__ Q, int heads, int T,
                             int total) {
  int i = blockIdx.x * blockDim.x + threadIdx.x;
  if (i >= total) return;
  int d = i & 63;
  int tmp = i >> 6;
  int h = tmp % heads;
  int row = tmp / heads;
  int t = row % T;
  float freq = (d < 32) ? powf(10000.0f, -(float)d / 31.0f) : 0.0f;
  float theta = (float)t * freq;
  float c = cosf(theta), s = sinf(theta);
  size_t base = (size_t)row * heads * 128 + (size_t)h * 128 + d;
  float x1 = (float)Q[base];
  float x2 = (float)Q[base + 64];
  Q[base]      = (__bf16)(x1 * c + x2 * s);
  Q[base + 64] = (__bf16)(-x1 * s + x2 * c);
}

// ---------------------------------------------------------------------------
// Flash attention, causal, GQA (16 Q heads share 4 KV heads).
// One wave per 16-query tile; 32-key steps; online softmax; all matmuls WMMA.
// V pre-transposed so P@V B-fragments are contiguous b128 loads.
__global__ __launch_bounds__(256) void flash_attn(
    const __bf16* __restrict__ Q, const __bf16* __restrict__ Km,
    const __bf16* __restrict__ Vt, __bf16* __restrict__ O, int T) {
  __shared__ __bf16 lP[8][16 * 32];  // per-wave P staging (S->A-frag transpose)
  int wave = threadIdx.x >> 5;
  int lane = threadIdx.x & 31;
  int hf = lane >> 4, idx = lane & 15;

  int qtiles = T / 16;
  int task = blockIdx.x * 8 + wave;  // (b, h, qt)
  int b = task / (16 * qtiles);
  int rem = task % (16 * qtiles);
  int h = rem / qtiles;
  int qt = rem % qtiles;
  int q0 = qt * 16;
  int kh = h >> 2;  // N_REP = 4
  const int ldq = 2048, ldkv = 512, Dk = 128;
  const float scale = 0.08838834764831845f;  // 1/sqrt(128)

  size_t qbase = ((size_t)b * T + q0) * ldq + (size_t)h * Dk;
  size_t kbase = (size_t)b * T * ldkv + (size_t)kh * Dk;
  size_t vtbase = (size_t)((b * 4 + kh) * 128) * T;

  v16bf qf[4];
#pragma unroll
  for (int dc = 0; dc < 4; ++dc)
    qf[dc] = load_a_frag(Q + qbase + dc * 32, ldq);

  v8f o[8] = {};
  float mstate[8], lstate[8], alpha[8];
#pragma unroll
  for (int i = 0; i < 8; ++i) { mstate[i] = -1e30f; lstate[i] = 0.0f; }

  __bf16* Plds = lP[wave];

  for (int k0 = 0; k0 <= q0 + 15; k0 += 32) {
    // S = Q @ K^T for keys [k0, k0+32): two 16-key groups
    v8f sa = {}, sb = {};
#pragma unroll
    for (int dc = 0; dc < 4; ++dc) {
      v16bf kb0 = load_b_frag(Km + kbase + (size_t)k0 * ldkv + dc * 32, ldkv);
      sa = wmma_bf16(qf[dc], kb0, sa);
      v16bf kb1 =
          load_b_frag(Km + kbase + (size_t)(k0 + 16) * ldkv + dc * 32, ldkv);
      sb = wmma_bf16(qf[dc], kb1, sb);
    }

    // scale + causal mask + online softmax (rows live in 16-lane groups)
#pragma unroll
    for (int i = 0; i < 8; ++i) {
      int m = q0 + i + hf * 8;  // absolute query index of this row
      float va = (k0 + idx <= m) ? sa[i] * scale : -1e30f;
      float vb = (k0 + 16 + idx <= m) ? sb[i] * scale : -1e30f;
      float rm = fmaxf(va, vb);
      rm = fmaxf(rm, __shfl_xor(rm, 1));
      rm = fmaxf(rm, __shfl_xor(rm, 2));
      rm = fmaxf(rm, __shfl_xor(rm, 4));
      rm = fmaxf(rm, __shfl_xor(rm, 8));
      float mnew = fmaxf(mstate[i], rm);
      alpha[i] = __expf(mstate[i] - mnew);
      float pa = __expf(va - mnew);
      float pb = __expf(vb - mnew);
      float rs = pa + pb;
      rs += __shfl_xor(rs, 1);
      rs += __shfl_xor(rs, 2);
      rs += __shfl_xor(rs, 4);
      rs += __shfl_xor(rs, 8);
      lstate[i] = lstate[i] * alpha[i] + rs;
      mstate[i] = mnew;
      int ml = i + hf * 8;
      Plds[ml * 32 + idx] = (__bf16)pa;
      Plds[ml * 32 + 16 + idx] = (__bf16)pb;
    }

    // rescale running O by alpha (per row)
#pragma unroll
    for (int f = 0; f < 8; ++f)
#pragma unroll
      for (int i = 0; i < 8; ++i) o[f][i] *= alpha[i];

    // LDS RAW within the wave: wait for ds stores before transposed reload
    asm volatile("s_wait_dscnt 0x0" ::: "memory");
    v16bf pf = load_a_frag(Plds, 32);

    // O += P(16x32) @ V(32x128): 8 d-blocks of 16, B-frags from Vt rows
#pragma unroll
    for (int f = 0; f < 8; ++f) {
      v16bf vb = load_b_frag(Vt + vtbase + (size_t)(f * 16) * T + k0, T);
      o[f] = wmma_bf16(pf, vb, o[f]);
    }
    asm volatile("" ::: "memory");
  }

  // normalize and write out (B*T, 2048) at head offset
#pragma unroll
  for (int f = 0; f < 8; ++f)
#pragma unroll
    for (int i = 0; i < 8; ++i) {
      int m = i + hf * 8;
      float val = o[f][i] / lstate[i];
      O[((size_t)b * T + q0 + m) * 2048 + (size_t)h * 128 + f * 16 + idx] =
          (__bf16)val;
    }
}

// ---------------------------------------------------------------------------
extern "C" void kernel_launch(void* const* d_in, const int* in_sizes, int n_in,
                              void* d_out, int out_size, void* d_ws,
                              size_t ws_size, hipStream_t stream) {
  const int B = 2, T = 2048, D = 2048, H = 16, KV = 4, DK = 128;
  const int M = B * T;   // 4096
  const int NQ = H * DK; // 2048
  const int NKV = KV * DK; // 512

  const float* x = (const float*)d_in[0];
  const float* wq = (const float*)d_in[1];
  const float* wk = (const float*)d_in[2];
  const float* wv = (const float*)d_in[3];
  const float* wo = (const float*)d_in[4];

  __bf16* p = (__bf16*)d_ws;
  __bf16* xbf = p;  p += (size_t)M * D;
  __bf16* wqb = p;  p += (size_t)NQ * D;
  __bf16* wkb = p;  p += (size_t)NKV * D;
  __bf16* wvb = p;  p += (size_t)NKV * D;
  __bf16* wob = p;  p += (size_t)D * D;
  __bf16* Qb  = p;  p += (size_t)M * NQ;
  __bf16* Kb  = p;  p += (size_t)M * NKV;
  __bf16* Vb  = p;  p += (size_t)M * NKV;
  __bf16* Vtb = p;  p += (size_t)M * NKV;
  __bf16* Ab  = p;  p += (size_t)M * D;

  auto cast = [&](const float* src, __bf16* dst, int n) {
    cast_f32_bf16<<<(n + 255) / 256, 256, 0, stream>>>(src, dst, n);
  };
  cast(x,  xbf, M * D);
  cast(wq, wqb, NQ * D);
  cast(wk, wkb, NKV * D);
  cast(wv, wvb, NKV * D);
  cast(wo, wob, D * D);

  // projections (bf16 WMMA, bf16 out)
  gemm_xwT<false><<<dim3(M / 128, NQ / 128), 256, 0, stream>>>(xbf, wqb, Qb, M,
                                                               NQ, D);
  gemm_xwT<false><<<dim3(M / 128, NKV / 128), 256, 0, stream>>>(xbf, wkb, Kb,
                                                                M, NKV, D);
  gemm_xwT<false><<<dim3(M / 128, NKV / 128), 256, 0, stream>>>(xbf, wvb, Vb,
                                                                M, NKV, D);

  // RoPE on Q and K (in place)
  {
    int nq = M * H * 64;
    rope_inplace<<<(nq + 255) / 256, 256, 0, stream>>>(Qb, H, T, nq);
    int nk = M * KV * 64;
    rope_inplace<<<(nk + 255) / 256, 256, 0, stream>>>(Kb, KV, T, nk);
  }

  // V transpose for contiguous P@V fragments
  {
    int total = B * KV * DK * T;  // 2,097,152
    transpose_v<<<(total + 255) / 256, 256, 0, stream>>>(Vb, Vtb, T, total);
  }

  // flash attention: B*H*(T/16) wave-tasks, 8 waves/block
  {
    int tasks = B * H * (T / 16);  // 4096
    flash_attn<<<tasks / 8, 256, 0, stream>>>(Qb, Kb, Vtb, Ab, T);
  }

  // output projection -> f32 d_out
  gemm_xwT<true><<<dim3(M / 128, D / 128), 256, 0, stream>>>(Ab, wob, d_out, M,
                                                             D, D);
}
